// ProbMPS_33655363731633
// MI455X (gfx1250) — matrix operand order repaired and checked by reading, not compiled, not run on hardware
//
#include <hip/hip_runtime.h>
#include <math.h>

#define SEQ   256
#define INP   4
#define BOND  128
#define BATCH 2048

#define AS       132   // LDS row stride (bf16 elems) for 128-wide bf16 tiles (padded)
#define AF       132   // LDS row stride (f32 elems) for TDM-loaded f32 tiles (pad 4 dw / 128 dw)
#define HS       33    // LDS row stride (f32 elems) for 32-wide H tile (padded)
#define AMP_COLS 32    // batch columns per amp workgroup (2 N-tiles)
#define ZWG      INP   // Z-scan cooperating workgroups (one per token)

typedef __attribute__((ext_vector_type(16))) __bf16 v16bf;
typedef __attribute__((ext_vector_type(8)))  float  v8f;
typedef __attribute__((ext_vector_type(4)))  unsigned int u32x4;
typedef __attribute__((ext_vector_type(8)))  unsigned int u32x8;

__device__ __forceinline__ v8f vzero8() {
  v8f z;
#pragma unroll
  for (int i = 0; i < 8; ++i) z[i] = 0.f;
  return z;
}

__device__ __forceinline__ v8f wmma_bf16(v16bf a, v16bf b, v8f c) {
  // D(f32 16x16) = A(bf16 16x32) x B(bf16 32x16) + C
  return __builtin_amdgcn_wmma_f32_16x16x32_bf16(false, a, false, b, (short)0, c,
                                                 false, false);
}

// =====================================================================
// Tensor Data Mover: DMA one 128x128 f32 tile global->LDS with 4-dword
// padding after every 128 dwords (row stride in LDS becomes AF=132 f32).
// D# layout per CDNA5 ISA ch.8 (group0 128b, group1 256b, groups2/3 zero).
// Issued per-wave (TDM ignores EXEC); caller gates to one wave per WG.
// =====================================================================
__device__ __forceinline__ void tdm_load_tile_f32(unsigned int lds_off,
                                                  const void* gsrc) {
  unsigned long long ga = (unsigned long long)(__SIZE_TYPE__)gsrc;
  u32x4 g0;
  g0[0] = 1u;                                              // count=1, user D#
  g0[1] = lds_off;                                         // lds_addr (bytes)
  g0[2] = (unsigned int)ga;                                // global_addr[31:0]
  g0[3] = (unsigned int)((ga >> 32) & 0x01ffffffu)         // global_addr[56:32]
        | (2u << 30);                                      // type=2 ("image")
  u32x8 g1;
  g1[0] = (2u << 16)      // data_size = 4B
        | (1u << 20)      // pad_enable
        | (6u << 22)      // pad_interval: every 128 dwords
        | (3u << 25);     // pad_amount: 4 dwords
  g1[1] = ((unsigned)BOND & 0xffffu) << 16;   // tensor_dim0.lo16 (=128)
  g1[2] = ((unsigned)BOND & 0xffffu) << 16;   // tensor_dim1.lo16 (=128)
  g1[3] = ((unsigned)BOND & 0xffffu) << 16;   // tile_dim0 (=128)
  g1[4] = (unsigned)BOND;                     // tile_dim1 (=128), tile_dim2=0
  g1[5] = (unsigned)BOND;                     // tensor_dim0_stride[31:0] (=128)
  g1[6] = 0u;                                 // stride0.hi16, stride1.lo16
  g1[7] = 0u;
  u32x4 g2 = {0u, 0u, 0u, 0u};
  u32x4 g3 = {0u, 0u, 0u, 0u};
  asm volatile("tensor_load_to_lds %0, %1, %2, %3"
               :: "s"(g0), "s"(g1), "s"(g2), "s"(g3)
               : "memory");
}

// ---- fragment builders (wave32); element s of lane half h holds k = 16h + s ----
// A operand: 16x32 bf16 from bf16 LDS tile (row stride AS).
__device__ __forceinline__ v16bf frag_a(const __bf16* s, int rowBase, int kOff, int lane) {
  const int m = rowBase + (lane & 15);
  const int half = lane >> 4;
  const __bf16* rp = s + m * AS + kOff;
  v16bf a;
#pragma unroll
  for (int v = 0; v < 8; ++v) {
    int k = 2 * v + 16 * half;
    a[2 * v]     = rp[k];
    a[2 * v + 1] = rp[k + 1];
  }
  return a;
}

// A operand from TDM-staged f32 tile (row stride AF), cvt at read.
__device__ __forceinline__ v16bf frag_a_f32(const float* s, int rowBase, int kOff, int lane) {
  const int m = rowBase + (lane & 15);
  const int half = lane >> 4;
  const float* rp = s + m * AF + kOff;
  v16bf a;
#pragma unroll
  for (int v = 0; v < 8; ++v) {
    int k = 2 * v + 16 * half;
    a[2 * v]     = (__bf16)rp[k];
    a[2 * v + 1] = (__bf16)rp[k + 1];
  }
  return a;
}

// B operand from f32 LDS H tile [k][n], per-lane column predicate (token mask).
__device__ __forceinline__ v16bf frag_b_f32(const float* s, int colBase, int kOff,
                                            int lane, bool act) {
  const int n = colBase + (lane & 15);
  const int half = lane >> 4;
  v16bf b;
#pragma unroll
  for (int v = 0; v < 8; ++v) {
    int k = kOff + 2 * v + 16 * half;
    float x0 = act ? s[k * HS + n] : 0.f;
    float x1 = act ? s[(k + 1) * HS + n] : 0.f;
    b[2 * v]     = (__bf16)x0;
    b[2 * v + 1] = (__bf16)x1;
  }
  return b;
}

// B operand from bf16 LDS tile stored row-major [k][n].
__device__ __forceinline__ v16bf frag_b_bf16(const __bf16* s, int colBase, int kOff, int lane) {
  const int n = colBase + (lane & 15);
  const int half = lane >> 4;
  v16bf b;
#pragma unroll
  for (int v = 0; v < 8; ++v) {
    int k = kOff + 2 * v + 16 * half;
    b[2 * v]     = s[k * AS + n];
    b[2 * v + 1] = s[(k + 1) * AS + n];
  }
  return b;
}

// B operand = A^T, i.e. B[k][n] = A[n][k] read from TDM-staged f32 A tile.
__device__ __forceinline__ v16bf frag_bT_f32(const float* s, int colBase, int kOff, int lane) {
  const int n = colBase + (lane & 15);
  const int half = lane >> 4;
  const float* rp = s + n * AF + kOff;
  v16bf b;
#pragma unroll
  for (int v = 0; v < 8; ++v) {
    int k = 2 * v + 16 * half;
    b[2 * v]     = (__bf16)rp[k];
    b[2 * v + 1] = (__bf16)rp[k + 1];
  }
  return b;
}

// Device-scope sense-reversal barrier for the 4 cooperating Z workgroups.
__device__ __forceinline__ void global_barrier(int* cnt, int* gen, int nwg) {
  __syncthreads();
  if (threadIdx.x == 0) {
    __threadfence();
    int g = __hip_atomic_load(gen, __ATOMIC_ACQUIRE, __HIP_MEMORY_SCOPE_AGENT);
    int arrived = __hip_atomic_fetch_add(cnt, 1, __ATOMIC_ACQ_REL, __HIP_MEMORY_SCOPE_AGENT) + 1;
    if (arrived == nwg) {
      __hip_atomic_store(cnt, 0, __ATOMIC_RELAXED, __HIP_MEMORY_SCOPE_AGENT);
      __hip_atomic_fetch_add(gen, 1, __ATOMIC_RELEASE, __HIP_MEMORY_SCOPE_AGENT);
    } else {
      while (__hip_atomic_load(gen, __ATOMIC_ACQUIRE, __HIP_MEMORY_SCOPE_AGENT) == g) {
        __builtin_amdgcn_s_sleep(2);
      }
    }
    __threadfence();
  }
  __syncthreads();
}

// =====================================================================
// Amplitude scan: h <- normalize(A_t[x_b] h), logn accumulation, final
// out[b] = 2*(log|h.omega| + logn[b]) - logZ  (logZ read from workspace).
// A slabs are double-buffered via TDM: wave0 issues slab u+1, waits
// TENSORcnt<=1 so slab u is complete while u+1 streams in.
// Dynamic LDS: two 128x132 f32 A buffers.
// =====================================================================
__global__ __launch_bounds__(256) void mps_amp_scan(
    const int* __restrict__ tok, const float* __restrict__ core,
    const float* __restrict__ edge, const float* __restrict__ wsf,
    float* __restrict__ out) {
  extern __shared__ char dsm[];
  float* sA0 = (float*)dsm;
  float* sA1 = sA0 + BOND * AF;
  __shared__ float  sH[BOND * HS];      // resident state, f32
  __shared__ float  sLogn[AMP_COLS];
  __shared__ int    sTok[AMP_COLS];

  const int tid = threadIdx.x;
  const int lane = tid & 31;
  const int wid = tid >> 5;             // wave id: owns M-tile [16*wid, 16*wid+16)
  const int half = lane >> 4;
  const int colBase = blockIdx.x * AMP_COLS;

  const unsigned int ldsA0 = (unsigned int)(__SIZE_TYPE__)(void*)sA0;
  const unsigned int ldsA1 = (unsigned int)(__SIZE_TYPE__)(void*)sA1;

  // h0 = alpha broadcast to all columns
  for (int idx = tid; idx < BOND * AMP_COLS; idx += 256) {
    int r = idx >> 5;
    int c = idx & 31;
    sH[r * HS + c] = edge[r];
  }
  if (tid < AMP_COLS) sLogn[tid] = 0.f;

  if (wid == 0)  // pipeline prologue: slab u=0 -> buffer 0
    tdm_load_tile_f32(ldsA0, core);
  __syncthreads();

  for (int t = 0; t < SEQ; ++t) {
    if (tid < AMP_COLS) sTok[tid] = tok[t * BATCH + colBase + tid];
    v8f acc0 = vzero8();
    v8f acc1 = vzero8();

    for (int i = 0; i < INP; ++i) {
      const int u = t * INP + i;
      if (wid == 0) {
        if (u + 1 < SEQ * INP) {
          tdm_load_tile_f32(((u + 1) & 1) ? ldsA1 : ldsA0,
                            core + (size_t)(u + 1) * (BOND * BOND));
          __builtin_amdgcn_s_wait_tensorcnt(1);  // slab u complete, u+1 in flight
        } else {
          __builtin_amdgcn_s_wait_tensorcnt(0);
        }
      }
      __syncthreads();   // slab u visible to all waves; tokens visible at i==0

      const float* sAcur = (u & 1) ? sA1 : sA0;
      const bool act0 = (sTok[(lane & 15)] == i);
      const bool act1 = (sTok[16 + (lane & 15)] == i);
#pragma unroll
      for (int kk = 0; kk < 4; ++kk) {
        v16bf a  = frag_a_f32(sAcur, 16 * wid, 32 * kk, lane);
        v16bf b0 = frag_b_f32(sH, 0, 32 * kk, lane, act0);
        acc0 = wmma_bf16(a, b0, acc0);
        v16bf b1 = frag_b_f32(sH, 16, 32 * kk, lane, act1);
        acc1 = wmma_bf16(a, b1, acc1);
      }
      __syncthreads();   // reads of slab u done (buffer reused at u+2)
    }

#pragma unroll
    for (int v = 0; v < 8; ++v) {
      int r = 16 * wid + v + 8 * half;
      sH[r * HS + (lane & 15)]      = acc0[v];
      sH[r * HS + 16 + (lane & 15)] = acc1[v];
    }
    __syncthreads();

    if (tid < AMP_COLS) {   // per-column renorm + log accumulation
      float ss = 0.f;
      for (int r = 0; r < BOND; ++r) { float x = sH[r * HS + tid]; ss += x * x; }
      float s = sqrtf(ss);
      sLogn[tid] += logf(s);
      float inv = 1.f / s;
      for (int r = 0; r < BOND; ++r) sH[r * HS + tid] *= inv;
    }
    __syncthreads();
  }

  if (tid < AMP_COLS) {
    float a = 0.f;
    for (int r = 0; r < BOND; ++r) a += sH[r * HS + tid] * edge[BOND + r];
    out[colBase + tid] = 2.f * (logf(fabsf(a)) + sLogn[tid]) - wsf[0];
  }
}

// =====================================================================
// Z scan: M <- (1/s) * sum_i A_i M A_i^T, logz += log s; 4 WGs (one per
// token) cooperate via a global barrier; result logZ -> wsf[0].
// A_t[wg] staged per step by TDM, overlapped with M->bf16 LDS staging.
// Workspace floats: [0]=logZ [1]=inv_s ; Mcur @ +64 floats; P_i slabs follow.
// Dynamic LDS: sAf (128x132 f32, TDM) + sB (128x132 bf16: M then T).
// =====================================================================
__global__ __launch_bounds__(256) void mps_z_scan(
    const float* __restrict__ core, const float* __restrict__ edge,
    float* __restrict__ wsf, int* __restrict__ bar) {
  extern __shared__ char zsm[];
  float*  sAf = (float*)zsm;                                 // A_t[wg], f32 (TDM)
  __bf16* sB  = (__bf16*)(zsm + BOND * AF * sizeof(float));  // M, then T (bf16)
  __shared__ float sRed[256];

  const int tid = threadIdx.x;
  const int lane = tid & 31;
  const int wid = tid >> 5;
  const int half = lane >> 4;
  const int wg = blockIdx.x;
  float* Mcur  = wsf + 64;
  float* Pmine = Mcur + BOND * BOND * (1 + wg);
  const unsigned int ldsA = (unsigned int)(__SIZE_TYPE__)(void*)sAf;

  if (wg == 0) {
    for (int idx = tid; idx < BOND * BOND; idx += 256)
      Mcur[idx] = edge[idx >> 7] * edge[idx & 127];   // M0 = alpha alpha^T
    if (tid == 0) wsf[1] = 1.f;
  }
  float logz = 0.f;
  global_barrier(bar, bar + 1, ZWG);

  for (int t = 0; t < SEQ; ++t) {
    // DMA this step's A tile; overlaps with the M staging below.
    if (wid == 0)
      tdm_load_tile_f32(ldsA, core + (size_t)(t * INP + wg) * (BOND * BOND));

    const float inv_s = wsf[1];
    for (int idx = tid; idx < BOND * BOND; idx += 256)
      sB[(idx >> 7) * AS + (idx & 127)] = (__bf16)(Mcur[idx] * inv_s);

    if (wid == 0) __builtin_amdgcn_s_wait_tensorcnt(0);
    __syncthreads();   // M staged everywhere; A tile landed

    // T = M x A^T  (each wave: one 16-row tile across all 8 column tiles)
    v8f accT[8];
#pragma unroll
    for (int nt = 0; nt < 8; ++nt) accT[nt] = vzero8();
#pragma unroll
    for (int kk = 0; kk < 4; ++kk) {
      v16bf m = frag_a(sB, 16 * wid, 32 * kk, lane);
#pragma unroll
      for (int nt = 0; nt < 8; ++nt) {
        v16bf at = frag_bT_f32(sAf, 16 * nt, 32 * kk, lane);
        accT[nt] = wmma_bf16(m, at, accT[nt]);
      }
    }
    __syncthreads();  // all M reads done; reuse sB for T
#pragma unroll
    for (int nt = 0; nt < 8; ++nt)
#pragma unroll
      for (int v = 0; v < 8; ++v)
        sB[(16 * wid + v + 8 * half) * AS + 16 * nt + (lane & 15)] = (__bf16)accT[nt][v];
    __syncthreads();

    // P = A x T -> this WG's global partial slab
#pragma unroll 1
    for (int nt = 0; nt < 8; ++nt) {
      v8f acc = vzero8();
#pragma unroll
      for (int kk = 0; kk < 4; ++kk) {
        v16bf a  = frag_a_f32(sAf, 16 * wid, 32 * kk, lane);
        v16bf bt = frag_b_bf16(sB, 16 * nt, 32 * kk, lane);
        acc = wmma_bf16(a, bt, acc);
      }
#pragma unroll
      for (int v = 0; v < 8; ++v)
        Pmine[(16 * wid + v + 8 * half) * BOND + 16 * nt + (lane & 15)] = acc[v];
    }
    global_barrier(bar, bar + 1, ZWG);

    if (wg == 0) {  // Mnew = sum_i P_i ; Frobenius renorm; logz accumulation
      const float* P0 = Mcur + BOND * BOND;
      const float* P1 = P0 + BOND * BOND;
      const float* P2 = P1 + BOND * BOND;
      const float* P3 = P2 + BOND * BOND;
      float ss = 0.f;
      for (int idx = tid; idx < BOND * BOND; idx += 256) {
        float m = P0[idx] + P1[idx] + P2[idx] + P3[idx];
        Mcur[idx] = m;
        ss += m * m;
      }
      sRed[tid] = ss;
      __syncthreads();
      for (int off = 128; off > 0; off >>= 1) {
        if (tid < off) sRed[tid] += sRed[tid + off];
        __syncthreads();
      }
      float s = sqrtf(sRed[0]);
      logz += logf(s);
      if (tid == 0) wsf[1] = 1.f / s;
      __syncthreads();
    }
    global_barrier(bar, bar + 1, ZWG);
  }

  if (wg == 0) {  // logZ = log(omega^T M omega) + logz
    const float inv_s = wsf[1];
    float part = 0.f;
    for (int idx = tid; idx < BOND * BOND; idx += 256)
      part += edge[BOND + (idx >> 7)] * Mcur[idx] * edge[BOND + (idx & 127)];
    sRed[tid] = part * inv_s;
    __syncthreads();
    for (int off = 128; off > 0; off >>= 1) {
      if (tid < off) sRed[tid] += sRed[tid + off];
      __syncthreads();
    }
    if (tid == 0) wsf[0] = logf(sRed[0]) + logz;
  }
}

extern "C" void kernel_launch(void* const* d_in, const int* in_sizes, int n_in,
                              void* d_out, int out_size, void* d_ws, size_t ws_size,
                              hipStream_t stream) {
  (void)in_sizes; (void)n_in; (void)out_size; (void)ws_size;
  const int*   tok  = (const int*)d_in[0];    // [SEQ, BATCH] int32
  const float* core = (const float*)d_in[1];  // [SEQ, INP, BOND, BOND] f32
  const float* edge = (const float*)d_in[2];  // [2, BOND] f32
  float* out = (float*)d_out;                 // [BATCH] f32
  float* wsf = (float*)d_ws;
  int*   bar = (int*)((char*)d_ws + 64);

  // Zero logZ/inv_s scalars + barrier counters each launch (ws is poisoned once).
  hipMemsetAsync(d_ws, 0, 256, stream);

  const size_t zlds = (size_t)BOND * AF * sizeof(float) +
                      (size_t)BOND * AS * sizeof(__bf16);
  mps_z_scan<<<dim3(ZWG), dim3(256), zlds, stream>>>(core, edge, wsf, bar);

  const size_t alds = (size_t)2 * BOND * AF * sizeof(float);
  mps_amp_scan<<<dim3(BATCH / AMP_COLS), dim3(256), 0 + alds, stream>>>(tok, core, edge, wsf, out);
}
